// RVAE_25778393710846
// MI455X (gfx1250) — matrix operand order repaired and checked by reading, not compiled, not run on hardware
//
#include <hip/hip_runtime.h>
#include <hip/hip_bf16.h>

typedef __attribute__((ext_vector_type(16))) _Float16 v16h;
typedef __attribute__((ext_vector_type(8)))  _Float16 v8h;
typedef __attribute__((ext_vector_type(8)))  float    v8f;

#define Bdim 256
#define Hdim 256
#define Ldim 512
#define G3H  768
#define NTHR 256
#define NWG2 16           // one WG per 16-row batch block
#define MROWS 16
#define LROWS 384         // Whh rows resident in LDS (waves 0..3); rows >=384 stream from L2

// LDS layout (byte offsets into dynamic shared)
#define OFF_W    0
#define OFF_H16  (LROWS * 256 * 2)            // 196608
#define OFF_GH   (OFF_H16 + MROWS * 256 * 2)  // 204800
#define OFF_GX   (OFF_GH + MROWS * G3H * 4)   // 253952
#define OFF_H32  (OFF_GX + MROWS * 256 * 4)   // 270336
#define OFF_MISC (OFF_H32 + MROWS * 256 * 4)  // 286720
#define SMEM_BYTES (OFF_MISC + 2048)          // 288768 <= 320KB/WGP

// ---------------------------------------------------------------- helpers ---

__device__ __forceinline__ float sigmoidf_(float x) { return 1.f / (1.f + __expf(-x)); }
__device__ __forceinline__ float softplusf_(float x) { return (x > 20.f) ? x : __logf(1.f + __expf(x)); }

// 16-bit fragment (16x32): lanes 0-15 hold K=k0..k0+15,
// lanes 16-31 hold K=k0+16..31. Works for LDS or global source (inlined).
__device__ __forceinline__ v16h load_frag16(const _Float16* __restrict__ base, int sel) {
    v8h lo = *(const v8h*)(base + 2 * sel);
    v8h hi = *(const v8h*)(base + 2 * sel + 8);
    v16h r;
#pragma unroll
    for (int i = 0; i < 8; ++i) { r[i] = lo[i]; r[i + 8] = hi[i]; }
    return r;
}

// B fragment: lane j<16 -> col n0+j, K=k0..k0+15 ; lane>=16 -> col n0+j-16, K=k0+16..31
__device__ __forceinline__ v16h load_bfrag16(const _Float16* __restrict__ p) {
    v8h lo = *(const v8h*)(p);
    v8h hi = *(const v8h*)(p + 8);
    v16h r;
#pragma unroll
    for (int i = 0; i < 8; ++i) { r[i] = lo[i]; r[i + 8] = hi[i]; }
    return r;
}

// Stage rows [0, LROWS) of the f16 weight matrix (768x256) into LDS, 16B vector copies.
__device__ __forceinline__ void stage_ldsW(const _Float16* __restrict__ W16, _Float16* ldsW) {
    const int tid = threadIdx.x;
    const v8h* src = (const v8h*)W16;
    v8h* dst = (v8h*)ldsW;
    for (int i = tid; i < LROWS * 256 / 8; i += NTHR) dst[i] = src[i];
    __syncthreads();
}

// One timestep pre-activation GEMM for this WG's 16 batch rows, all 768 columns.
// Waves 0..3: B-frags from LDS; waves 4..7: B-frags streamed from global (L2).
// h-part always -> ch (gh). x-part: n-gate cols (>=512) -> cx (gix), else merged into ch.
template <int KX>
__device__ __forceinline__ void gemm_step(
    const _Float16* __restrict__ ldsW,    // LDS: LROWS x 256
    const _Float16* __restrict__ ldsH16,  // LDS: 16 x 256 (h, f16)
    const _Float16* __restrict__ Whh16,   // global f16: 768 x 256
    const _Float16* __restrict__ Wih16,   // global f16: 768 x KX
    const _Float16* __restrict__ axRow,   // x_t for this WG's rows: 16 x KX (global or LDS)
    float* __restrict__ ldsGh,            // LDS: 16 x 768
    float* __restrict__ ldsGx) {          // LDS: 16 x 256
    const int tid  = threadIdx.x;
    const int wave = tid >> 5;
    const int lane = tid & 31;
    const int lrow = lane & 15;
    const int sel  = (lane & 16) ? 8 : 0;
    const int koff = (lane & 16) ? 16 : 0;
    const int nloc0 = 96 * wave;

    v8f zero = {};
    v8f ch[6], cx[6];
#pragma unroll
    for (int t2 = 0; t2 < 6; ++t2) { ch[t2] = zero; cx[t2] = zero; }

    if (nloc0 < LROWS) {  // waves 0..3: LDS-resident weights
#pragma unroll 2
        for (int kk = 0; kk < 8; ++kk) {
            const int k0 = kk * 32;
            v16h A = load_frag16(ldsH16 + lrow * 256 + k0, sel);
#pragma unroll
            for (int t2 = 0; t2 < 6; ++t2) {
                const _Float16* bp = ldsW + (nloc0 + 16 * t2 + lrow) * 256 + k0 + koff;
                v16h Bf = load_bfrag16(bp);
                ch[t2] = __builtin_amdgcn_wmma_f32_16x16x32_f16(false, A, false, Bf,
                                                                (short)0, ch[t2], false, false);
            }
        }
    } else {              // waves 4..7: stream weights from L2
#pragma unroll 2
        for (int kk = 0; kk < 8; ++kk) {
            const int k0 = kk * 32;
            v16h A = load_frag16(ldsH16 + lrow * 256 + k0, sel);
#pragma unroll
            for (int t2 = 0; t2 < 6; ++t2) {
                const _Float16* bp = Whh16 + (nloc0 + 16 * t2 + lrow) * 256 + k0 + koff;
                v16h Bf = load_bfrag16(bp);
                ch[t2] = __builtin_amdgcn_wmma_f32_16x16x32_f16(false, A, false, Bf,
                                                                (short)0, ch[t2], false, false);
            }
        }
    }
    // input part: K = KX (64 or 32), weights streamed from L2
#pragma unroll
    for (int kk = 0; kk < KX / 32; ++kk) {
        const int k0 = kk * 32;
        v16h A = load_frag16(axRow + lrow * KX + k0, sel);
#pragma unroll
        for (int t2 = 0; t2 < 6; ++t2) {
            const int n0 = nloc0 + 16 * t2;
            const _Float16* bp = Wih16 + (n0 + lrow) * KX + k0 + koff;
            v16h Bf = load_bfrag16(bp);
            if (n0 >= 512)
                cx[t2] = __builtin_amdgcn_wmma_f32_16x16x32_f16(false, A, false, Bf,
                                                                (short)0, cx[t2], false, false);
            else
                ch[t2] = __builtin_amdgcn_wmma_f32_16x16x32_f16(false, A, false, Bf,
                                                                (short)0, ch[t2], false, false);
        }
    }
    // store C/D (ISA layout: vgpr r -> row r (+8 for lanes 16-31), col = n0 + (lane&15))
    const int rowoff = (lane & 16) ? 8 : 0;
#pragma unroll
    for (int t2 = 0; t2 < 6; ++t2) {
        const int n0  = nloc0 + 16 * t2;
        const int col = n0 + lrow;
#pragma unroll
        for (int r = 0; r < 8; ++r) {
            const int row = r + rowoff;
            ldsGh[row * G3H + col] = ch[t2][r];
            if (n0 >= 512) ldsGx[row * 256 + (col - 512)] = cx[t2][r];
        }
    }
}

// GRU gate update over this WG's 16x256 state; h carried in 16 regs per thread.
__device__ __forceinline__ void gate_update(
    const float* __restrict__ ldsGh, const float* __restrict__ ldsGx,
    const float* __restrict__ bih, const float* __restrict__ bhh,
    float* __restrict__ ldsH32, _Float16* __restrict__ ldsH16,
    float hreg[16], _Float16* __restrict__ xhr_out) {
    const int tid = threadIdx.x;
#pragma unroll 4
    for (int i = 0; i < 16; ++i) {
        const int e = tid + i * NTHR;
        const int b = e >> 8, j = e & 255;
        float r  = sigmoidf_(ldsGh[b * G3H + j] + bih[j] + bhh[j]);
        float zg = sigmoidf_(ldsGh[b * G3H + 256 + j] + bih[256 + j] + bhh[256 + j]);
        float n  = tanhf(ldsGx[b * 256 + j] + bih[512 + j] +
                         r * (ldsGh[b * G3H + 512 + j] + bhh[512 + j]));
        float hn = (1.f - zg) * n + zg * hreg[i];
        hreg[i] = hn;
        ldsH32[b * 256 + j] = hn;
        _Float16 hh = (_Float16)hn;
        ldsH16[b * 256 + j] = hh;
        if (xhr_out) xhr_out[b * 256 + j] = hh;
    }
}

// ---------------------------------------------------------------- kernels ---

__global__ void __launch_bounds__(256) f32_to_f16_kernel(const float* __restrict__ src,
                                                         _Float16* __restrict__ dst, int n) {
    int i = blockIdx.x * 256 + threadIdx.x;
    if (i < n) dst[i] = (_Float16)src[i];
}

// xrev16[t,b,k] = f16( tanh(x[b, L-1-t] * fx_W[k] + fx_b[k]) )
__global__ void __launch_bounds__(256) xf_kernel(const float* __restrict__ x,
                                                 const float* __restrict__ fxW,
                                                 const float* __restrict__ fxb,
                                                 _Float16* __restrict__ xrev16) {
    int idx = blockIdx.x * 256 + threadIdx.x;
    if (idx >= Ldim * Bdim * 64) return;
    int k = idx & 63;
    int b = (idx >> 6) & 255;
    int t = idx >> 14;
    float xv = x[b * Ldim + (Ldim - 1 - t)];
    xrev16[idx] = (_Float16)tanhf(xv * fxW[k] + fxb[k]);
}

__global__ void __launch_bounds__(NTHR, 1) scan_phi1(
    const _Float16* __restrict__ Whh16, const _Float16* __restrict__ Wih16,
    const float* __restrict__ bih, const float* __restrict__ bhh,
    const _Float16* __restrict__ xrev16, _Float16* __restrict__ xhr16) {
    extern __shared__ char smem[];
    _Float16* ldsW   = (_Float16*)(smem + OFF_W);
    _Float16* ldsH16 = (_Float16*)(smem + OFF_H16);
    float*    ldsGh  = (float*)(smem + OFF_GH);
    float*    ldsGx  = (float*)(smem + OFF_GX);
    float*    ldsH32 = (float*)(smem + OFF_H32);
    const int tid = threadIdx.x;
    const size_t rowbase = (size_t)blockIdx.x * MROWS;

    stage_ldsW(Whh16, ldsW);
    for (int i = tid; i < MROWS * 256; i += NTHR) ldsH16[i] = (_Float16)0.f;
    float hreg[16];
#pragma unroll
    for (int i = 0; i < 16; ++i) hreg[i] = 0.f;
    __syncthreads();

    for (int t = 0; t < Ldim; ++t) {
        gemm_step<64>(ldsW, ldsH16, Whh16, Wih16,
                      xrev16 + ((size_t)t * Bdim + rowbase) * 64, ldsGh, ldsGx);
        __syncthreads();
        gate_update(ldsGh, ldsGx, bih, bhh, ldsH32, ldsH16, hreg,
                    xhr16 + ((size_t)(Ldim - 1 - t) * Bdim + rowbase) * Hdim);
        if (t + 1 < Ldim)
            __builtin_prefetch(xrev16 + ((size_t)(t + 1) * Bdim + rowbase) * 64 + tid * 4, 0, 1);
        __syncthreads();
    }
}

__global__ void __launch_bounds__(NTHR, 1) scan_enc(
    const _Float16* __restrict__ Whh16, const _Float16* __restrict__ Wih16,
    const float* __restrict__ bih, const float* __restrict__ bhh,
    const _Float16* __restrict__ xhr16, _Float16* __restrict__ zfseq16,
    const float* __restrict__ pnLW, const float* __restrict__ pnLb,
    const float* __restrict__ pnSW, const float* __restrict__ pnSb,
    const float* __restrict__ eps,
    const float* __restrict__ fzW1, const float* __restrict__ fzb1,
    const float* __restrict__ fzW2, const float* __restrict__ fzb2) {
    extern __shared__ char smem[];
    _Float16* ldsW   = (_Float16*)(smem + OFF_W);
    _Float16* ldsH16 = (_Float16*)(smem + OFF_H16);
    float*    ldsGh  = (float*)(smem + OFF_GH);
    float*    ldsGx  = (float*)(smem + OFF_GX);
    float*    ldsH32 = (float*)(smem + OFF_H32);
    float*    ldsZ   = (float*)(smem + OFF_MISC);              // 64 floats
    _Float16* ldsZf  = (_Float16*)(smem + OFF_MISC + 256);     // 16 x 32 halves
    const int tid = threadIdx.x;
    const size_t rowbase = (size_t)blockIdx.x * MROWS;

    stage_ldsW(Whh16, ldsW);
    for (int i = tid; i < MROWS * 256; i += NTHR) ldsH16[i] = (_Float16)0.f;
    if (tid < MROWS * 32) ldsZf[tid] = (_Float16)0.f;  // tid<512 covered by NTHR? (512>256)
    if (tid + NTHR < MROWS * 32) ldsZf[tid + NTHR] = (_Float16)0.f;
    float hreg[16];
#pragma unroll
    for (int i = 0; i < 16; ++i) hreg[i] = 0.f;
    __syncthreads();

    for (int t = 0; t < Ldim; ++t) {
        gemm_step<32>(ldsW, ldsH16, Whh16, Wih16, ldsZf, ldsGh, ldsGx);
        __syncthreads();
        gate_update(ldsGh, ldsGx, bih, bhh, ldsH32, ldsH16, hreg, nullptr);
        __syncthreads();
        // latent projections: thread (b,zi), b local 0..15, zi 0..3
        if (tid < MROWS * 4) {
            const int b = tid >> 2, zi = tid & 3;
            const size_t bg = rowbase + b;
            float accL = pnLb[zi], accS = pnSb[zi];
            const float*    wl  = pnLW + zi * 512;
            const float*    wsc = pnSW + zi * 512;
            const _Float16* xr  = xhr16 + ((size_t)t * Bdim + bg) * Hdim;
            const float*    zr  = ldsH32 + b * 256;
            for (int j = 0; j < 256; ++j) {
                float zv = zr[j], xv = (float)xr[j];
                accL += zv * wl[j]  + xv * wl[256 + j];
                accS += zv * wsc[j] + xv * wsc[256 + j];
            }
            ldsZ[tid] = accL + softplusf_(accS) * eps[(size_t)t * Bdim * 4 + bg * 4 + zi];
        }
        __syncthreads();
        // feat_z: 4 -> 2 -> 32, one thread per local batch row
        if (tid < MROWS) {
            const int b = tid;
            float z0 = ldsZ[b * 4], z1 = ldsZ[b * 4 + 1], z2 = ldsZ[b * 4 + 2], z3 = ldsZ[b * 4 + 3];
            float fh0 = tanhf(fzW1[0] * z0 + fzW1[1] * z1 + fzW1[2] * z2 + fzW1[3] * z3 + fzb1[0]);
            float fh1 = tanhf(fzW1[4] * z0 + fzW1[5] * z1 + fzW1[6] * z2 + fzW1[7] * z3 + fzb1[1]);
            _Float16* zfo = ldsZf + b * 32;
            _Float16* zso = zfseq16 + ((size_t)t * Bdim + rowbase + b) * 32;
            for (int o = 0; o < 32; ++o) {
                float v = tanhf(fzW2[o * 2] * fh0 + fzW2[o * 2 + 1] * fh1 + fzb2[o]);
                _Float16 vh = (_Float16)v;
                zfo[o] = vh;
                zso[o] = vh;
            }
        }
        __syncthreads();   // ldsZf ready for next step's x-part A fragments
    }
}

__global__ void __launch_bounds__(NTHR, 1) scan_th(
    const _Float16* __restrict__ Whh16, const _Float16* __restrict__ Wih16,
    const float* __restrict__ bih, const float* __restrict__ bhh,
    const _Float16* __restrict__ zfseq16,
    const float* __restrict__ tnW, const float* __restrict__ tnb,
    float* __restrict__ out) {
    extern __shared__ char smem[];
    _Float16* ldsW   = (_Float16*)(smem + OFF_W);
    _Float16* ldsH16 = (_Float16*)(smem + OFF_H16);
    float*    ldsGh  = (float*)(smem + OFF_GH);
    float*    ldsGx  = (float*)(smem + OFF_GX);
    float*    ldsH32 = (float*)(smem + OFF_H32);
    const int tid = threadIdx.x;
    const size_t rowbase = (size_t)blockIdx.x * MROWS;

    stage_ldsW(Whh16, ldsW);
    for (int i = tid; i < MROWS * 256; i += NTHR) ldsH16[i] = (_Float16)0.f;
    float hreg[16];
#pragma unroll
    for (int i = 0; i < 16; ++i) hreg[i] = 0.f;
    __syncthreads();

    for (int t = 0; t < Ldim; ++t) {
        gemm_step<32>(ldsW, ldsH16, Whh16, Wih16,
                      zfseq16 + ((size_t)t * Bdim + rowbase) * 32, ldsGh, ldsGx);
        __syncthreads();
        gate_update(ldsGh, ldsGx, bih, bhh, ldsH32, ldsH16, hreg, nullptr);
        if (t + 1 < Ldim)
            __builtin_prefetch(zfseq16 + ((size_t)(t + 1) * Bdim + rowbase) * 32 + tid * 2, 0, 1);
        __syncthreads();
        // output head: x_loc[b, t] = h . tn_locW + tn_locb   (out is (B, L, 1))
        if (tid < MROWS) {
            const float* hr = ldsH32 + tid * 256;
            float acc = tnb[0];
            for (int j = 0; j < 256; ++j) acc += hr[j] * tnW[j];
            out[(rowbase + tid) * Ldim + t] = acc;
        }
    }
}

// ------------------------------------------------------------------ host ---

extern "C" void kernel_launch(void* const* d_in, const int* in_sizes, int n_in,
                              void* d_out, int out_size, void* d_ws, size_t ws_size,
                              hipStream_t stream) {
    (void)in_sizes; (void)n_in; (void)out_size; (void)ws_size;
    const float* x     = (const float*)d_in[0];
    const float* eps   = (const float*)d_in[1];
    const float* fxW   = (const float*)d_in[2];
    const float* fxb   = (const float*)d_in[3];
    const float* fzW1  = (const float*)d_in[4];
    const float* fzb1  = (const float*)d_in[5];
    const float* fzW2  = (const float*)d_in[6];
    const float* fzb2  = (const float*)d_in[7];
    const float* p1Wih = (const float*)d_in[8];
    const float* p1Whh = (const float*)d_in[9];
    const float* p1bih = (const float*)d_in[10];
    const float* p1bhh = (const float*)d_in[11];
    const float* p2Wih = (const float*)d_in[12];
    const float* p2Whh = (const float*)d_in[13];
    const float* p2bih = (const float*)d_in[14];
    const float* p2bhh = (const float*)d_in[15];
    const float* pnLW  = (const float*)d_in[16];
    const float* pnLb  = (const float*)d_in[17];
    const float* pnSW  = (const float*)d_in[18];
    const float* pnSb  = (const float*)d_in[19];
    const float* thWih = (const float*)d_in[20];
    const float* thWhh = (const float*)d_in[21];
    const float* thbih = (const float*)d_in[22];
    const float* thbhh = (const float*)d_in[23];
    const float* tnW   = (const float*)d_in[24];
    const float* tnb   = (const float*)d_in[25];
    float* out = (float*)d_out;

    char* cur = (char*)d_ws;
    auto alloc = [&](size_t bytes) -> char* {
        char* r = cur;
        cur += (bytes + 255) & ~(size_t)255;
        return r;
    };
    _Float16* xrev16   = (_Float16*)alloc((size_t)Ldim * Bdim * 64 * 2);
    _Float16* xhr16    = (_Float16*)alloc((size_t)Ldim * Bdim * Hdim * 2);
    _Float16* zfseq16  = (_Float16*)alloc((size_t)Ldim * Bdim * 32 * 2);
    _Float16* Whh1_16  = (_Float16*)alloc((size_t)G3H * Hdim * 2);
    _Float16* Wih1_16  = (_Float16*)alloc((size_t)G3H * 64 * 2);
    _Float16* Whh2_16  = (_Float16*)alloc((size_t)G3H * Hdim * 2);
    _Float16* Wih2_16  = (_Float16*)alloc((size_t)G3H * 32 * 2);
    _Float16* WhhT_16  = (_Float16*)alloc((size_t)G3H * Hdim * 2);
    _Float16* WihT_16  = (_Float16*)alloc((size_t)G3H * 32 * 2);

    // f32 -> f16 weight copies (streamed portions + LDS staging source)
    const int nWhh = G3H * Hdim, nWih1 = G3H * 64, nWih2 = G3H * 32;
    f32_to_f16_kernel<<<(nWhh + 255) / 256, 256, 0, stream>>>(p1Whh, Whh1_16, nWhh);
    f32_to_f16_kernel<<<(nWih1 + 255) / 256, 256, 0, stream>>>(p1Wih, Wih1_16, nWih1);
    f32_to_f16_kernel<<<(nWhh + 255) / 256, 256, 0, stream>>>(p2Whh, Whh2_16, nWhh);
    f32_to_f16_kernel<<<(nWih2 + 255) / 256, 256, 0, stream>>>(p2Wih, Wih2_16, nWih2);
    f32_to_f16_kernel<<<(nWhh + 255) / 256, 256, 0, stream>>>(thWhh, WhhT_16, nWhh);
    f32_to_f16_kernel<<<(nWih2 + 255) / 256, 256, 0, stream>>>(thWih, WihT_16, nWih2);

    const int nxf = Ldim * Bdim * 64;
    xf_kernel<<<(nxf + 255) / 256, 256, 0, stream>>>(x, fxW, fxb, xrev16);

    scan_phi1<<<NWG2, NTHR, SMEM_BYTES, stream>>>(Whh1_16, Wih1_16, p1bih, p1bhh,
                                                  xrev16, xhr16);

    scan_enc<<<NWG2, NTHR, SMEM_BYTES, stream>>>(Whh2_16, Wih2_16, p2bih, p2bhh,
                                                 xhr16, zfseq16,
                                                 pnLW, pnLb, pnSW, pnSb, eps,
                                                 fzW1, fzb1, fzW2, fzb2);

    scan_th<<<NWG2, NTHR, SMEM_BYTES, stream>>>(WhhT_16, WihT_16, thbih, thbhh,
                                                zfseq16, tnW, tnb, out);
}
